// AttentionDecoder_82738249990894
// MI455X (gfx1250) — matrix operand order repaired and checked by reading, not compiled, not run on hardware
//
#include <hip/hip_runtime.h>
#include <hip/hip_bf16.h>
#include <math.h>

#define B_ 4
#define T_ 1024
#define C_ 1024
#define H_ 16
#define D_ 64
#define F_ 4096
#define L_ 4
#define OUT_ 65

typedef __attribute__((ext_vector_type(16))) __bf16 v16bf;
typedef __attribute__((ext_vector_type(8)))  float v8f;

union FragBF { v16bf v; unsigned int u[8]; unsigned short s[16]; };

// Native fp32 -> bf16 (lowers to v_cvt_pk_bf16_f32 / v_cvt_bf16_f32, RTNE)
__device__ inline unsigned short f2bf(float f) {
    union { __bf16 h; unsigned short s; } c;
    c.h = (__bf16)f;
    return c.s;
}
__device__ inline unsigned pk2(float a, float b) {
    union { __bf16 h[2]; unsigned u; } c;
    c.h[0] = (__bf16)a; c.h[1] = (__bf16)b;
    return c.u;
}
__device__ inline v8f v8f_zero() {
    v8f z;
#pragma unroll
    for (int i = 0; i < 8; i++) z[i] = 0.0f;
    return z;
}

// A-fragment (16x32 bf16, MxK): lanes 0-15 = rows, VGPR0-3 K=0..7(+hi*16), VGPR4-7 K=8..15(+hi*16)
template<int LD>
__device__ inline v16bf lds_a_frag(const unsigned short* base, int row, int kk, int hi) {
    FragBF f;
    const unsigned short* p = base + row * LD + kk + hi * 16;
#pragma unroll
    for (int i = 0; i < 4; i++) f.u[i]     = *(const unsigned int*)(p + 2 * i);
#pragma unroll
    for (int i = 0; i < 4; i++) f.u[4 + i] = *(const unsigned int*)(p + 8 + 2 * i);
    return f.v;
}
// B-fragment (32x16 bf16, KxN), stored transposed in LDS as [col][k]:
// lanes 0-15 hold K=0..15, lanes 16-31 hold K=16..31 for column N=lane&15
template<int LD>
__device__ inline v16bf lds_b_frag(const unsigned short* base, int col, int kk, int hi) {
    FragBF f;
    const unsigned short* p = base + col * LD + kk + hi * 16;
#pragma unroll
    for (int i = 0; i < 8; i++) f.u[i] = *(const unsigned int*)(p + 2 * i);
    return f.v;
}

// ---------------------------------------------------------------------------
// x[b,t,c] = concat(emb[acts], durations) + pos
// ---------------------------------------------------------------------------
__global__ __launch_bounds__(256) void embed_kernel(
    const int* __restrict__ acts, const float* __restrict__ dur,
    const float* __restrict__ emb, const float* __restrict__ pos,
    float* __restrict__ x)
{
    int row = blockIdx.x;              // b*T + t
    int t = row % T_;
    int a = acts[row];
    for (int c = threadIdx.x; c < C_; c += 256) {
        float v = (c < C_ - 1) ? emb[(long)a * (C_ - 1) + c] : dur[row];
        x[(long)row * C_ + c] = v + pos[(long)t * C_ + c];
    }
}

// ---------------------------------------------------------------------------
// h = x * rsqrt(mean(x^2) + eps) * g   (one block per row, C=1024)
// ---------------------------------------------------------------------------
__global__ __launch_bounds__(256) void rmsnorm_kernel(
    const float* __restrict__ x, const float* __restrict__ g, float* __restrict__ h)
{
    __shared__ float sb[256];
    int row = blockIdx.x;
    const float* xr = x + (long)row * C_;
    float s = 0.0f;
    for (int c = threadIdx.x; c < C_; c += 256) { float v = xr[c]; s += v * v; }
    sb[threadIdx.x] = s;
    __syncthreads();
    for (int st = 128; st > 0; st >>= 1) {
        if (threadIdx.x < st) sb[threadIdx.x] += sb[threadIdx.x + st];
        __syncthreads();
    }
    float rs = rsqrtf(sb[0] * (1.0f / (float)C_) + 1.1920929e-07f);
    for (int c = threadIdx.x; c < C_; c += 256)
        h[(long)row * C_ + c] = xr[c] * rs * g[c];
}

// ---------------------------------------------------------------------------
// Repack Wq/Wk/Wv for one layer: (H,C,D) -> (C, H*D) so QKV are plain GEMMs
// ---------------------------------------------------------------------------
__global__ __launch_bounds__(256) void repack_qkv_kernel(
    const float* __restrict__ Wq, const float* __restrict__ Wk, const float* __restrict__ Wv,
    float* __restrict__ pq, float* __restrict__ pk, float* __restrict__ pv)
{
    long idx = (long)blockIdx.x * 256 + threadIdx.x;   // c*1024 + (h*64+d)
    int c = (int)(idx >> 10);
    int n = (int)(idx & 1023);
    int hh = n >> 6, d = n & 63;
    long src = ((long)hh * C_ + c) * D_ + d;
    pq[idx] = Wq[src]; pk[idx] = Wk[src]; pv[idx] = Wv[src];
}

// ---------------------------------------------------------------------------
// Tiled bf16-WMMA GEMM: Out[M,N] = epilogue(A[M,K] @ B[K,N])
//   EPI 0: plain   1: +bias   2: +bias +residual   3: gelu(x+bias) exact
// Block = 256 thr (8 waves), 128x128 tile, K staged 32 wide with double-buffered
// LDS: global_load_b128 next tile || v_wmma on current tile || cvt+ds_store.
// ---------------------------------------------------------------------------
template<int EPI>
__global__ __launch_bounds__(256) void gemm_bf16(
    const float* __restrict__ A, int lda,
    const float* __restrict__ Bm, int ldb,
    float* __restrict__ Out, int ldo,
    const float* __restrict__ bias,
    const float* __restrict__ Res, int ldr,
    int M, int N, int K)
{
    __shared__ unsigned short As[2][128 * 40];   // [row][k], pad 40 vs bank conflicts
    __shared__ unsigned short Bs[2][128 * 40];   // transposed: [col][k]

    int tid  = threadIdx.x;
    int lane = tid & 31, wave = tid >> 5;
    int low  = lane & 15, hi = lane >> 4;
    int wm = wave >> 1, wn = wave & 1;           // 4x2 wave grid -> 32x64 per wave
    int bm0 = blockIdx.y * 128, bn0 = blockIdx.x * 128;

    v8f acc[2][4];
#pragma unroll
    for (int i = 0; i < 2; i++)
#pragma unroll
        for (int j = 0; j < 4; j++) acc[i][j] = v8f_zero();

    int arow = tid >> 1, akoff = (tid & 1) * 16;   // A: 2 thr/row, 16 fp32 each
    int bkrow = tid >> 3, bcoff = (tid & 7) * 16;  // B: 8 thr/row(k), 16 cols each

    float4 aR0, aR1, aR2, aR3;
    float  bR[16];

    auto gload = [&](int k0) {
        int grow = bm0 + arow;
        if (grow < M) {
            const float4* ap = (const float4*)(A + (long)grow * lda + k0 + akoff);
            aR0 = ap[0]; aR1 = ap[1]; aR2 = ap[2]; aR3 = ap[3];
        } else {
            aR0 = aR1 = aR2 = aR3 = make_float4(0.f, 0.f, 0.f, 0.f);
        }
        int gk = k0 + bkrow;
        const float* bp = Bm + (long)gk * ldb + bn0 + bcoff;
        if (bn0 + bcoff + 16 <= N && (ldb & 3) == 0) {
            const float4* bv = (const float4*)bp;
            float4 f0 = bv[0], f1 = bv[1], f2 = bv[2], f3 = bv[3];
            bR[0]=f0.x; bR[1]=f0.y; bR[2]=f0.z;  bR[3]=f0.w;
            bR[4]=f1.x; bR[5]=f1.y; bR[6]=f1.z;  bR[7]=f1.w;
            bR[8]=f2.x; bR[9]=f2.y; bR[10]=f2.z; bR[11]=f2.w;
            bR[12]=f3.x;bR[13]=f3.y;bR[14]=f3.z; bR[15]=f3.w;
        } else {
#pragma unroll
            for (int i = 0; i < 16; i++) {
                int col = bn0 + bcoff + i;
                bR[i] = (col < N) ? bp[i] : 0.0f;
            }
        }
    };
    auto sstore = [&](int buf) {
        unsigned* dst = (unsigned*)&As[buf][arow * 40 + akoff];
        dst[0] = pk2(aR0.x, aR0.y); dst[1] = pk2(aR0.z, aR0.w);
        dst[2] = pk2(aR1.x, aR1.y); dst[3] = pk2(aR1.z, aR1.w);
        dst[4] = pk2(aR2.x, aR2.y); dst[5] = pk2(aR2.z, aR2.w);
        dst[6] = pk2(aR3.x, aR3.y); dst[7] = pk2(aR3.z, aR3.w);
#pragma unroll
        for (int i = 0; i < 16; i++)
            Bs[buf][(bcoff + i) * 40 + bkrow] = f2bf(bR[i]);
    };
    auto compute = [&](int buf) {
        v16bf bfr[4];
#pragma unroll
        for (int nt = 0; nt < 4; nt++)
            bfr[nt] = lds_b_frag<40>(Bs[buf], wn * 64 + nt * 16 + low, 0, hi);
#pragma unroll
        for (int mt = 0; mt < 2; mt++) {
            v16bf a = lds_a_frag<40>(As[buf], wm * 32 + mt * 16 + low, 0, hi);
#pragma unroll
            for (int nt = 0; nt < 4; nt++)
                acc[mt][nt] = __builtin_amdgcn_wmma_f32_16x16x32_bf16(
                    false, a, false, bfr[nt], (short)0, acc[mt][nt], false, false);
        }
    };

    int nk = K >> 5;                 // K is always a multiple of 32 here
    gload(0);
    sstore(0);
    __syncthreads();
    for (int t = 0; t < nk; ++t) {
        if (t + 1 < nk) gload((t + 1) * 32);   // global loads in flight...
        compute(t & 1);                        // ...while matrix pipe works
        if (t + 1 < nk) sstore((t + 1) & 1);   // cvt + stage into other buffer
        __syncthreads();
    }

    // ---- epilogue: C/D layout lanes 0-15 -> M=r, lanes 16-31 -> M=r+8, N=lane&15
    int orow = bm0 + wm * 32;
    int ocol = bn0 + wn * 64;
#pragma unroll
    for (int mt = 0; mt < 2; mt++)
#pragma unroll
        for (int nt = 0; nt < 4; nt++)
#pragma unroll
            for (int r = 0; r < 8; r++) {
                int grow = orow + mt * 16 + r + hi * 8;
                int gcol = ocol + nt * 16 + low;
                if (grow < M && gcol < N) {
                    float v = acc[mt][nt][r];
                    if constexpr (EPI >= 1) v += bias[gcol];
                    if constexpr (EPI == 3) v = 0.5f * v * (1.0f + erff(v * 0.70710678118654752f));
                    if constexpr (EPI == 2) v += Res[(long)grow * ldr + gcol];
                    Out[(long)grow * ldo + gcol] = v;
                }
            }
}

// ---------------------------------------------------------------------------
// Flash-style causal attention, one (b, head, 64-query-block) per block.
// S = Q K^T and O += P V both via v_wmma_f32_16x16x32_bf16; online softmax in LDS.
// q/k/v layout: [B,T,H*D] row-major (col = h*64+d). o written same layout.
// ---------------------------------------------------------------------------
__global__ __launch_bounds__(256) void attn_kernel(
    const float* __restrict__ Q, const float* __restrict__ K,
    const float* __restrict__ V, float* __restrict__ O)
{
    __shared__ unsigned short Qb[64 * 72];   // [row][d]
    __shared__ unsigned short Kb[64 * 72];   // [key][d]
    __shared__ unsigned short Vt[64 * 72];   // transposed: [d][key]
    __shared__ unsigned short Pb[64 * 72];   // probs bf16 [row][key]
    __shared__ float Sf[64 * 68];            // scores fp32
    __shared__ float red[64 * 4];
    __shared__ float rmax[64], rsum[64], ralpha[64];

    int tid  = threadIdx.x;
    int lane = tid & 31, wave = tid >> 5;
    int low  = lane & 15, hi = lane >> 4;
    int wm = wave >> 1, wn = wave & 1;        // wave -> 16 rows x 32 cols (2 tiles)
    int qblk = blockIdx.x;
    int bh = blockIdx.y;
    int b = bh / H_, hh = bh % H_;
    int qbase = qblk * 64;

    int lr = tid >> 2, ld0 = (tid & 3) * 16;  // loader mapping: row, 16-wide d chunk
    const float scale = 0.125f;               // D^-0.5

    // preload Q tile (fp32 -> bf16)
    {
        const float4* qp = (const float4*)(Q + ((long)(b * T_ + qbase + lr)) * (H_ * D_) + hh * D_ + ld0);
        float4 f0 = qp[0], f1 = qp[1], f2 = qp[2], f3 = qp[3];
        unsigned* dst = (unsigned*)&Qb[lr * 72 + ld0];
        dst[0] = pk2(f0.x, f0.y); dst[1] = pk2(f0.z, f0.w);
        dst[2] = pk2(f1.x, f1.y); dst[3] = pk2(f1.z, f1.w);
        dst[4] = pk2(f2.x, f2.y); dst[5] = pk2(f2.z, f2.w);
        dst[6] = pk2(f3.x, f3.y); dst[7] = pk2(f3.z, f3.w);
    }
    if (tid < 64) { rmax[tid] = -INFINITY; rsum[tid] = 0.0f; }

    v8f accO[2] = { v8f_zero(), v8f_zero() };
    __syncthreads();

    for (int j = 0; j <= qblk; ++j) {
        // ---- load K, V key-block (64 keys x 64 dims)
        {
            long rb = ((long)(b * T_ + j * 64 + lr)) * (H_ * D_) + hh * D_ + ld0;
            const float4* kp = (const float4*)(K + rb);
            float4 f0 = kp[0], f1 = kp[1], f2 = kp[2], f3 = kp[3];
            unsigned* dk = (unsigned*)&Kb[lr * 72 + ld0];
            dk[0] = pk2(f0.x, f0.y); dk[1] = pk2(f0.z, f0.w);
            dk[2] = pk2(f1.x, f1.y); dk[3] = pk2(f1.z, f1.w);
            dk[4] = pk2(f2.x, f2.y); dk[5] = pk2(f2.z, f2.w);
            dk[6] = pk2(f3.x, f3.y); dk[7] = pk2(f3.z, f3.w);
            const float4* vp = (const float4*)(V + rb);
            float4 g0 = vp[0], g1 = vp[1], g2 = vp[2], g3 = vp[3];
            float vv[16] = { g0.x, g0.y, g0.z, g0.w, g1.x, g1.y, g1.z, g1.w,
                             g2.x, g2.y, g2.z, g2.w, g3.x, g3.y, g3.z, g3.w };
#pragma unroll
            for (int i = 0; i < 16; i++) Vt[(ld0 + i) * 72 + lr] = f2bf(vv[i]);
        }
        __syncthreads();

        // ---- S = Q K^T (contraction over D=64 -> two k-steps of 32)
        v8f accS[2] = { v8f_zero(), v8f_zero() };
#pragma unroll
        for (int kk = 0; kk < 64; kk += 32) {
            v16bf a = lds_a_frag<72>(Qb, wm * 16 + low, kk, hi);
#pragma unroll
            for (int nt = 0; nt < 2; nt++) {
                v16bf bfr = lds_b_frag<72>(Kb, wn * 32 + nt * 16 + low, kk, hi);
                accS[nt] = __builtin_amdgcn_wmma_f32_16x16x32_bf16(
                    false, a, false, bfr, (short)0, accS[nt], false, false);
            }
        }
        // scale + causal mask -> Sf
#pragma unroll
        for (int nt = 0; nt < 2; nt++) {
            int scol = wn * 32 + nt * 16 + low;
            int gk = j * 64 + scol;
#pragma unroll
            for (int r = 0; r < 8; r++) {
                int srow = wm * 16 + r + hi * 8;
                int gq = qbase + srow;
                Sf[srow * 68 + scol] = (gk <= gq) ? accS[nt][r] * scale : -INFINITY;
            }
        }
        __syncthreads();

        // ---- online softmax: block max per row (4 threads/row x 16 cols)
        {
            int r = tid >> 2, sg = tid & 3;
            float m = -INFINITY;
#pragma unroll
            for (int c = 0; c < 16; c++) m = fmaxf(m, Sf[r * 68 + sg * 16 + c]);
            red[r * 4 + sg] = m;
        }
        __syncthreads();
        if (tid < 64) {
            float bm = fmaxf(fmaxf(red[tid * 4], red[tid * 4 + 1]),
                             fmaxf(red[tid * 4 + 2], red[tid * 4 + 3]));
            float mo = rmax[tid];
            float mn = fmaxf(mo, bm);
            ralpha[tid] = __expf(mo - mn);
            rmax[tid] = mn;
        }
        __syncthreads();
        {
            int r = tid >> 2, sg = tid & 3;
            float mn = rmax[r];
            float s = 0.0f;
#pragma unroll
            for (int c = 0; c < 16; c++) {
                float p = __expf(Sf[r * 68 + sg * 16 + c] - mn);
                Pb[r * 72 + sg * 16 + c] = f2bf(p);
                s += p;
            }
            red[r * 4 + sg] = s;
        }
        __syncthreads();
        if (tid < 64)
            rsum[tid] = rsum[tid] * ralpha[tid]
                      + red[tid * 4] + red[tid * 4 + 1] + red[tid * 4 + 2] + red[tid * 4 + 3];

        // ---- rescale O, then O += P V
#pragma unroll
        for (int nt = 0; nt < 2; nt++)
#pragma unroll
            for (int r = 0; r < 8; r++) {
                float al = ralpha[wm * 16 + r + hi * 8];
                accO[nt][r] = accO[nt][r] * al;
            }
#pragma unroll
        for (int kk = 0; kk < 64; kk += 32) {
            v16bf a = lds_a_frag<72>(Pb, wm * 16 + low, kk, hi);
#pragma unroll
            for (int nt = 0; nt < 2; nt++) {
                v16bf bfr = lds_b_frag<72>(Vt, wn * 32 + nt * 16 + low, kk, hi);
                accO[nt] = __builtin_amdgcn_wmma_f32_16x16x32_bf16(
                    false, a, false, bfr, (short)0, accO[nt], false, false);
            }
        }
        __syncthreads();
    }

    // ---- finalize: O / rowsum -> [B,T,H*D]
#pragma unroll
    for (int nt = 0; nt < 2; nt++)
#pragma unroll
        for (int r = 0; r < 8; r++) {
            int srow = wm * 16 + r + hi * 8;
            int d = wn * 32 + nt * 16 + low;
            float v = accO[nt][r] / rsum[srow];
            O[((long)(b * T_ + qbase + srow)) * (H_ * D_) + hh * D_ + d] = v;
        }
}

// ---------------------------------------------------------------------------
// head: log_softmax over 64 act logits + log(sigmoid(dur)); one row per block
// ---------------------------------------------------------------------------
__global__ __launch_bounds__(64) void head_kernel(
    const float* __restrict__ logits, float* __restrict__ out)
{
    __shared__ float buf[64];
    int r = blockIdx.x, t = threadIdx.x;
    float v = logits[(long)r * OUT_ + t];
    buf[t] = v;
    __syncthreads();
    for (int s = 32; s > 0; s >>= 1) { if (t < s) buf[t] = fmaxf(buf[t], buf[t + s]); __syncthreads(); }
    float m = buf[0];
    __syncthreads();
    buf[t] = __expf(v - m);
    __syncthreads();
    for (int s = 32; s > 0; s >>= 1) { if (t < s) buf[t] += buf[t + s]; __syncthreads(); }
    float lse = m + __logf(buf[0]);
    out[(long)r * OUT_ + t] = v - lse;
    if (t == 0) {
        float d = logits[(long)r * OUT_ + 64];
        out[(long)r * OUT_ + 64] = -log1pf(__expf(-d));   // log(sigmoid(d))
    }
}

// ---------------------------------------------------------------------------
extern "C" void kernel_launch(void* const* d_in, const int* in_sizes, int n_in,
                              void* d_out, int out_size, void* d_ws, size_t ws_size,
                              hipStream_t stream)
{
    (void)in_sizes; (void)n_in; (void)out_size; (void)ws_size;
    const int*   acts      = (const int*)d_in[0];
    const float* durations = (const float*)d_in[1];
    const float* emb_table = (const float*)d_in[2];
    const float* pos_table = (const float*)d_in[3];
    const float* Wq = (const float*)d_in[4];
    const float* Wk = (const float*)d_in[5];
    const float* Wv = (const float*)d_in[6];
    const float* Wo = (const float*)d_in[7];
    const float* bo = (const float*)d_in[8];
    const float* W1 = (const float*)d_in[9];
    const float* b1 = (const float*)d_in[10];
    const float* W2 = (const float*)d_in[11];
    const float* b2 = (const float*)d_in[12];
    const float* g1 = (const float*)d_in[13];
    const float* g2 = (const float*)d_in[14];
    const float* lm_W = (const float*)d_in[15];
    const float* lm_b = (const float*)d_in[16];

    char* ws = (char*)d_ws;
    const long MB = 1024L * 1024L;
    float* x      = (float*)(ws + 0 * MB);     // [4096,1024] 16MB
    float* h      = (float*)(ws + 16 * MB);    // [4096,1024] 16MB
    float* q      = (float*)(ws + 32 * MB);    // [4096,1024] 16MB
    float* k      = (float*)(ws + 48 * MB);    // 16MB
    float* v      = (float*)(ws + 64 * MB);    // 16MB
    float* o      = (float*)(ws + 80 * MB);    // 16MB
    float* mid    = (float*)(ws + 96 * MB);    // [4096,4096] 64MB
    float* pq     = (float*)(ws + 160 * MB);   // 4MB packed per-layer qkv weights
    float* pk     = (float*)(ws + 164 * MB);
    float* pv     = (float*)(ws + 168 * MB);
    float* logits = (float*)(ws + 172 * MB);   // [4096,65]

    const int M = B_ * T_;

    embed_kernel<<<M, 256, 0, stream>>>(acts, durations, emb_table, pos_table, x);

    dim3 gC((C_ + 127) / 128, (M + 127) / 128);
    dim3 gF((F_ + 127) / 128, (M + 127) / 128);

    for (int l = 0; l < L_; ++l) {
        rmsnorm_kernel<<<M, 256, 0, stream>>>(x, g1 + l * C_, h);
        repack_qkv_kernel<<<(C_ * H_ * D_) / 256, 256, 0, stream>>>(
            Wq + (long)l * H_ * C_ * D_, Wk + (long)l * H_ * C_ * D_,
            Wv + (long)l * H_ * C_ * D_, pq, pk, pv);

        gemm_bf16<0><<<gC, 256, 0, stream>>>(h, C_, pq, C_, q, C_, nullptr, nullptr, 0, M, C_, C_);
        gemm_bf16<0><<<gC, 256, 0, stream>>>(h, C_, pk, C_, k, C_, nullptr, nullptr, 0, M, C_, C_);
        gemm_bf16<0><<<gC, 256, 0, stream>>>(h, C_, pv, C_, v, C_, nullptr, nullptr, 0, M, C_, C_);

        attn_kernel<<<dim3(T_ / 64, B_ * H_), 256, 0, stream>>>(q, k, v, o);

        // x = x + o @ Wo + bo
        gemm_bf16<2><<<gC, 256, 0, stream>>>(o, C_, Wo + (long)l * C_ * C_, C_,
                                             x, C_, bo + l * C_, x, C_, M, C_, C_);
        rmsnorm_kernel<<<M, 256, 0, stream>>>(x, g2 + l * C_, h);
        // mid = gelu(h @ W1 + b1)
        gemm_bf16<3><<<gF, 256, 0, stream>>>(h, C_, W1 + (long)l * C_ * F_, F_,
                                             mid, F_, b1 + l * F_, nullptr, 0, M, F_, C_);
        // x = x + mid @ W2 + b2
        gemm_bf16<2><<<gC, 256, 0, stream>>>(mid, F_, W2 + (long)l * F_ * C_, C_,
                                             x, C_, b2 + l * C_, x, C_, M, C_, F_);
    }

    dim3 gO((OUT_ + 127) / 128, (M + 127) / 128);
    gemm_bf16<1><<<gO, 256, 0, stream>>>(x, C_, lm_W, OUT_, logits, OUT_,
                                         lm_b, nullptr, 0, M, OUT_, C_);
    head_kernel<<<M, 64, 0, stream>>>(logits, (float*)d_out);
}
